// TransformerBlock_1632087572649
// MI455X (gfx1250) — matrix operand order repaired and checked by reading, not compiled, not run on hardware
//
#include <hip/hip_runtime.h>
#include <hip/hip_bf16.h>

// ---------------------------------------------------------------------------
// Transformer block for MI455X (gfx1250): bf16 WMMA everywhere, fp32 accum.
// ---------------------------------------------------------------------------

typedef __attribute__((ext_vector_type(16))) __bf16 v16bf;
typedef __attribute__((ext_vector_type(8)))  float  v8f;

constexpr int Bb   = 4;
constexpr int Ss   = 2048;
constexpr int Dd   = 1024;
constexpr int Hh   = 16;
constexpr int Dhd  = 64;
constexpr int Ff   = 4096;
constexpr int Mtok = Bb * Ss; // 8192 tokens

union FragU { v16bf v; uint4 q[2]; };

// Two 16-byte loads forming one 16x32 bf16 fragment; `second` is the element
// offset of the second 8-element chunk (A and B frags: 16).
static __device__ __forceinline__ v16bf ld2(const __hip_bfloat16* p, int second) {
  FragU f;
  f.q[0] = *(const uint4*)p;
  f.q[1] = *(const uint4*)(p + second);
  return f.v;
}

// A fragment (16x32 bf16, row-major source, ld in elements).
// Lane L: row = m0 + (L&15); elems 0..7 -> K = k0+8*hi+0..7 ; 8..15 -> K = k0+16+8*hi+0..7
static __device__ __forceinline__ v16bf load_a_frag(const __hip_bfloat16* A, int ld,
                                                    int m0, int k0) {
  int lane = threadIdx.x & 31;
  int r = lane & 15, hi = lane >> 4;
  return ld2(A + (long)(m0 + r) * ld + k0 + 8 * hi, 16);
}

// B fragment (32x16 bf16) loaded from BT stored [N][K] row-major (ld = K).
// Lane L: col = n0 + (L&15); elems 0..7 -> K = k0+8*hi+0..7 ; 8..15 -> K = k0+16+8*hi+0..7
static __device__ __forceinline__ v16bf load_bt_frag(const __hip_bfloat16* Bt, int ld,
                                                     int n0, int k0) {
  int lane = threadIdx.x & 31;
  int c = lane & 15, hi = lane >> 4;
  return ld2(Bt + (long)(n0 + c) * ld + k0 + 8 * hi, 16);
}

static __device__ __forceinline__ v8f wmma_bf16(v16bf a, v16bf b, v8f c) {
  return __builtin_amdgcn_wmma_f32_16x16x32_bf16(false, a, false, b, (short)0, c,
                                                 false, false);
}

// ---------------------------------------------------------------------------
// Weight prep: fp32 [K][N] -> bf16 transposed [N][K]
// ---------------------------------------------------------------------------
__global__ __launch_bounds__(256) void transpose_to_bf16(const float* __restrict__ W,
                                                         __hip_bfloat16* __restrict__ Wt,
                                                         int K, int N) {
  __shared__ float tile[32][33];
  int kt = blockIdx.x * 32, nt = blockIdx.y * 32;
  int tx = threadIdx.x & 31, ty = threadIdx.x >> 5; // 8 rows of 32
  #pragma unroll
  for (int i = ty; i < 32; i += 8)
    tile[i][tx] = W[(long)(kt + i) * N + nt + tx];
  __syncthreads();
  #pragma unroll
  for (int i = ty; i < 32; i += 8)
    Wt[(long)(nt + i) * K + kt + tx] = __float2bfloat16(tile[tx][i]);
}

// ---------------------------------------------------------------------------
// LayerNorm (unbiased var, divide by sqrt(var)+eps) -> bf16 output
// ---------------------------------------------------------------------------
__global__ __launch_bounds__(256) void ln_kernel(const float* __restrict__ x,
                                                 const float* __restrict__ g,
                                                 const float* __restrict__ be,
                                                 __hip_bfloat16* __restrict__ out) {
  __shared__ float sh0[256], sh1[256];
  long row = blockIdx.x;
  const float* xr = x + row * Dd;
  float s = 0.f, ss = 0.f;
  for (int i = threadIdx.x; i < Dd; i += 256) {
    float v = xr[i];
    s += v; ss += v * v;
  }
  sh0[threadIdx.x] = s; sh1[threadIdx.x] = ss;
  __syncthreads();
  for (int off = 128; off > 0; off >>= 1) {
    if ((int)threadIdx.x < off) {
      sh0[threadIdx.x] += sh0[threadIdx.x + off];
      sh1[threadIdx.x] += sh1[threadIdx.x + off];
    }
    __syncthreads();
  }
  float mean = sh0[0] * (1.0f / Dd);
  float var  = (sh1[0] - (float)Dd * mean * mean) * (1.0f / (Dd - 1));
  float inv  = 1.0f / (sqrtf(var) + 1e-5f);
  for (int i = threadIdx.x; i < Dd; i += 256)
    out[row * Dd + i] = __float2bfloat16(g[i] * (xr[i] - mean) * inv + be[i]);
}

// ---------------------------------------------------------------------------
// Generic bf16 WMMA GEMM: C[M,N] = A[M,K] * BT[N,K]^T, fused epilogues.
// Block: 256 threads = 8 waves as 2(M) x 4(N); wave tile 64x32; block 128x128.
// K-loop is 2x-unrolled ping-pong software pipelined: two independent
// fragment register sets (no rotation movs). Fragment addresses are fixed
// per-lane bases (derived directly from the kernel args so address-space
// inference keeps the loads on the GLOBAL path, not FLAT) plus a loop-carried
// integer k offset; constant intra-iteration offsets fold into immediates.
// Requires K % 64 == 0 and K >= 128 (holds: K = 1024 or 4096).
// ---------------------------------------------------------------------------
enum { EP_QK = 0, EP_VT = 1, EP_ADDRES = 2, EP_GELU = 3 };

__global__ __launch_bounds__(256) void gemm_bf16_kernel(
    const __hip_bfloat16* __restrict__ A, const __hip_bfloat16* __restrict__ Bt,
    int N, int K, int mode,
    const float* __restrict__ bias, const float* __restrict__ resid,
    float* __restrict__ outF, __hip_bfloat16* __restrict__ outB) {
  int wave = threadIdx.x >> 5;
  int wm = wave >> 2, wn = wave & 3;
  int lane = threadIdx.x & 31;
  int c = lane & 15, r = lane & 15, hi = lane >> 4;
  int mBase = blockIdx.x * 128 + wm * 64;
  int n0    = blockIdx.y * 128 + wn * 32;

  v8f acc[4][2];
  #pragma unroll
  for (int t = 0; t < 4; ++t)
    #pragma unroll
    for (int u = 0; u < 2; ++u) acc[t][u] = (v8f)0.0f;

  // fixed per-lane fragment base pointers (k = 0), straight off kernel args
  const __hip_bfloat16* pB[2];
  const __hip_bfloat16* pA[4];
  #pragma unroll
  for (int u = 0; u < 2; ++u)
    pB[u] = Bt + (long)(n0 + u * 16 + c) * K + 8 * hi;
  #pragma unroll
  for (int t = 0; t < 4; ++t)
    pA[t] = A + (long)(mBase + t * 16 + r) * K + 8 * hi;

  v16bf b0[2], a0[4], b1[2], a1[4];
  #pragma unroll
  for (int u = 0; u < 2; ++u) b0[u] = ld2(pB[u], 16);
  #pragma unroll
  for (int t = 0; t < 4; ++t) a0[t] = ld2(pA[t], 16);

  for (int k0 = 0; k0 < K - 64; k0 += 64) {
    // prefetch k0+32 into set1
    #pragma unroll
    for (int u = 0; u < 2; ++u) b1[u] = ld2(pB[u] + k0 + 32, 16);
    #pragma unroll
    for (int t = 0; t < 4; ++t) a1[t] = ld2(pA[t] + k0 + 32, 16);
    // math on set0 (k0)
    #pragma unroll
    for (int t = 0; t < 4; ++t)
      #pragma unroll
      for (int u = 0; u < 2; ++u)
        acc[t][u] = wmma_bf16(a0[t], b0[u], acc[t][u]);
    // prefetch k0+64 into set0
    #pragma unroll
    for (int u = 0; u < 2; ++u) b0[u] = ld2(pB[u] + k0 + 64, 16);
    #pragma unroll
    for (int t = 0; t < 4; ++t) a0[t] = ld2(pA[t] + k0 + 64, 16);
    // math on set1 (k0+32)
    #pragma unroll
    for (int t = 0; t < 4; ++t)
      #pragma unroll
      for (int u = 0; u < 2; ++u)
        acc[t][u] = wmma_bf16(a1[t], b1[u], acc[t][u]);
  }
  // tail: set0 holds k = K-64; load K-32 into set1, finish both
  #pragma unroll
  for (int u = 0; u < 2; ++u) b1[u] = ld2(pB[u] + K - 32, 16);
  #pragma unroll
  for (int t = 0; t < 4; ++t) a1[t] = ld2(pA[t] + K - 32, 16);
  #pragma unroll
  for (int t = 0; t < 4; ++t)
    #pragma unroll
    for (int u = 0; u < 2; ++u)
      acc[t][u] = wmma_bf16(a0[t], b0[u], acc[t][u]);
  #pragma unroll
  for (int t = 0; t < 4; ++t)
    #pragma unroll
    for (int u = 0; u < 2; ++u)
      acc[t][u] = wmma_bf16(a1[t], b1[u], acc[t][u]);

  float bval[2] = {0.0f, 0.0f};
  if (mode == EP_ADDRES || mode == EP_GELU) {
    bval[0] = bias[n0 + c];
    bval[1] = bias[n0 + 16 + c];
  }

  #pragma unroll
  for (int t = 0; t < 4; ++t) {
    #pragma unroll
    for (int u = 0; u < 2; ++u) {
      #pragma unroll
      for (int i = 0; i < 8; ++i) {
        int m = mBase + t * 16 + i + 8 * hi;   // global row (token)
        int n = n0 + u * 16 + c;               // global col
        float v = acc[t][u][i];
        if (mode == EP_QK) {
          int b = m >> 11, s = m & (Ss - 1);
          int h = n >> 6,  dh = n & 63;
          outB[(((long)(b * Hh + h) * Ss) + s) * Dhd + dh] = __float2bfloat16(v);
        } else if (mode == EP_VT) {
          int b = m >> 11, s = m & (Ss - 1);
          int h = n >> 6,  dh = n & 63;
          outB[(((long)(b * Hh + h) * Dhd) + dh) * Ss + s] = __float2bfloat16(v);
        } else if (mode == EP_GELU) {
          float uu = v + bval[u];
          float gl = 0.5f * uu * (1.0f + erff(uu * 0.70710678118654752f));
          outB[(long)m * N + n] = __float2bfloat16(gl);
        } else { // EP_ADDRES
          outF[(long)m * N + n] = v + bval[u] + resid[(long)m * N + n];
        }
      }
    }
  }
}

// ---------------------------------------------------------------------------
// Causal flash attention: one wave per 16-query tile.
// Q,K: [B*H][S][64] bf16 ; Vt: [B*H][64][S] bf16 ; O: [B][S][D] bf16
// ---------------------------------------------------------------------------
__global__ __launch_bounds__(256) void attn_kernel(
    const __hip_bfloat16* __restrict__ Q, const __hip_bfloat16* __restrict__ Km,
    const __hip_bfloat16* __restrict__ Vt, __hip_bfloat16* __restrict__ O) {
  __shared__ __hip_bfloat16 ldsP[8][16][32];
  int wave = threadIdx.x >> 5;
  int lane = threadIdx.x & 31;
  int c = lane & 15, hi = lane >> 4;
  int h = blockIdx.y, b = blockIdx.z;
  int qtile = blockIdx.x * 8 + wave;
  int qbase = qtile * 16;
  long bh = (long)(b * Hh + h);
  const __hip_bfloat16* Qb = Q  + bh * Ss * Dhd;
  const __hip_bfloat16* Kb = Km + bh * Ss * Dhd;
  const __hip_bfloat16* Vb = Vt + bh * Dhd * Ss;

  v16bf qa0 = load_a_frag(Qb, Dhd, qbase, 0);
  v16bf qa1 = load_a_frag(Qb, Dhd, qbase, 32);

  v8f o[4];
  #pragma unroll
  for (int t = 0; t < 4; ++t) o[t] = (v8f)0.0f;
  float mrow[8], lrow[8];
  #pragma unroll
  for (int i = 0; i < 8; ++i) { mrow[i] = -1e30f; lrow[i] = 0.0f; }

  for (int kb = 0; kb < qbase + 16; kb += 32) {
    // ---- scores for 32 keys (two 16-key tiles), K-dim = Dh = 64 ----
    v8f st[2];
    #pragma unroll
    for (int t = 0; t < 2; ++t) {
      st[t] = (v8f)0.0f;
      v16bf kf0 = load_bt_frag(Kb, Dhd, kb + t * 16, 0);
      v16bf kf1 = load_bt_frag(Kb, Dhd, kb + t * 16, 32);
      st[t] = wmma_bf16(qa0, kf0, st[t]);
      st[t] = wmma_bf16(qa1, kf1, st[t]);
    }
    // ---- online softmax per row (row = i + 8*hi; cols across 16 lanes) ----
    #pragma unroll
    for (int i = 0; i < 8; ++i) {
      int row = i + 8 * hi;
      float s0 = st[0][i] * 0.125f;
      float s1 = st[1][i] * 0.125f;
      s0 = (kb + c      > qbase + row) ? -1e30f : s0;
      s1 = (kb + 16 + c > qbase + row) ? -1e30f : s1;
      float rm = fmaxf(s0, s1);
      #pragma unroll
      for (int off = 1; off < 16; off <<= 1)
        rm = fmaxf(rm, __shfl_xor(rm, off, 16));
      float mnew = fmaxf(mrow[i], rm);
      float p0 = __expf(s0 - mnew);
      float p1 = __expf(s1 - mnew);
      float rs = p0 + p1;
      #pragma unroll
      for (int off = 1; off < 16; off <<= 1)
        rs += __shfl_xor(rs, off, 16);
      float scale = __expf(mrow[i] - mnew);
      lrow[i] = lrow[i] * scale + rs;
      mrow[i] = mnew;
      #pragma unroll
      for (int t = 0; t < 4; ++t) o[t][i] *= scale;
      ldsP[wave][row][c]      = __float2bfloat16(p0);
      ldsP[wave][row][16 + c] = __float2bfloat16(p1);
    }
    // C-layout -> A-layout re-layout goes through LDS (cross-lane): wait for DS
    asm volatile("s_wait_dscnt 0" ::: "memory");
    FragU pf;
    {
      int r = lane & 15;
      pf.q[0] = *(const uint4*)&ldsP[wave][r][8 * hi];
      pf.q[1] = *(const uint4*)&ldsP[wave][r][16 + 8 * hi];
    }
    // ---- P(16x32) x V(32x64): 4 WMMAs, B from Vt (contiguous in key dim) ----
    #pragma unroll
    for (int t = 0; t < 4; ++t) {
      v16bf vf = load_bt_frag(Vb, Ss, t * 16, kb);
      o[t] = wmma_bf16(pf.v, vf, o[t]);
    }
  }

  // ---- normalize, write attn output [B][S][D] bf16 ----
  #pragma unroll
  for (int i = 0; i < 8; ++i) {
    int row = i + 8 * hi;
    float inv = 1.0f / lrow[i];
    long tok = (long)b * Ss + qbase + row;
    #pragma unroll
    for (int t = 0; t < 4; ++t)
      O[tok * Dd + h * Dhd + t * 16 + c] = __float2bfloat16(o[t][i] * inv);
  }
}

// ---------------------------------------------------------------------------
// Host launcher
// ---------------------------------------------------------------------------
extern "C" void kernel_launch(void* const* d_in, const int* in_sizes, int n_in,
                              void* d_out, int out_size, void* d_ws, size_t ws_size,
                              hipStream_t stream) {
  (void)in_sizes; (void)n_in; (void)out_size; (void)ws_size;
  const float* x   = (const float*)d_in[0];
  const float* Wq  = (const float*)d_in[1];
  const float* Wk  = (const float*)d_in[2];
  const float* Wv  = (const float*)d_in[3];
  const float* Wo  = (const float*)d_in[4];
  const float* bo  = (const float*)d_in[5];
  const float* W1  = (const float*)d_in[6];
  const float* b1  = (const float*)d_in[7];
  const float* W2  = (const float*)d_in[8];
  const float* b2  = (const float*)d_in[9];
  const float* g1  = (const float*)d_in[10];
  const float* be1 = (const float*)d_in[11];
  const float* g2  = (const float*)d_in[12];
  const float* be2 = (const float*)d_in[13];

  char* ws = (char*)d_ws;
  auto alloc = [&](size_t bytes) {
    char* p = ws;
    ws += (bytes + 255) & ~(size_t)255;
    return p;
  };
  __hip_bfloat16* zb   = (__hip_bfloat16*)alloc((size_t)Mtok * Dd * 2);
  __hip_bfloat16* WqT  = (__hip_bfloat16*)alloc((size_t)Dd * Dd * 2);
  __hip_bfloat16* WkT  = (__hip_bfloat16*)alloc((size_t)Dd * Dd * 2);
  __hip_bfloat16* WvT  = (__hip_bfloat16*)alloc((size_t)Dd * Dd * 2);
  __hip_bfloat16* WoT  = (__hip_bfloat16*)alloc((size_t)Dd * Dd * 2);
  __hip_bfloat16* W1T  = (__hip_bfloat16*)alloc((size_t)Ff * Dd * 2);
  __hip_bfloat16* W2T  = (__hip_bfloat16*)alloc((size_t)Dd * Ff * 2);
  __hip_bfloat16* Qbuf = (__hip_bfloat16*)alloc((size_t)Mtok * Dd * 2);
  __hip_bfloat16* Kbuf = (__hip_bfloat16*)alloc((size_t)Mtok * Dd * 2);
  __hip_bfloat16* Vtb  = (__hip_bfloat16*)alloc((size_t)Mtok * Dd * 2);
  __hip_bfloat16* attO = (__hip_bfloat16*)alloc((size_t)Mtok * Dd * 2);
  float*          zadd = (float*)alloc((size_t)Mtok * Dd * 4);
  __hip_bfloat16* znb  = (__hip_bfloat16*)alloc((size_t)Mtok * Dd * 2);
  __hip_bfloat16* hbuf = (__hip_bfloat16*)alloc((size_t)Mtok * Ff * 2);

  dim3 tb(256);
  // Weight prep (bf16 + transpose to [N][K])
  transpose_to_bf16<<<dim3(Dd / 32, Dd / 32), tb, 0, stream>>>(Wq, WqT, Dd, Dd);
  transpose_to_bf16<<<dim3(Dd / 32, Dd / 32), tb, 0, stream>>>(Wk, WkT, Dd, Dd);
  transpose_to_bf16<<<dim3(Dd / 32, Dd / 32), tb, 0, stream>>>(Wv, WvT, Dd, Dd);
  transpose_to_bf16<<<dim3(Dd / 32, Dd / 32), tb, 0, stream>>>(Wo, WoT, Dd, Dd);
  transpose_to_bf16<<<dim3(Dd / 32, Ff / 32), tb, 0, stream>>>(W1, W1T, Dd, Ff);
  transpose_to_bf16<<<dim3(Ff / 32, Dd / 32), tb, 0, stream>>>(W2, W2T, Ff, Dd);

  // z = LN1(x) (bf16)
  ln_kernel<<<Mtok, tb, 0, stream>>>(x, g1, be1, zb);

  // Q,K,V projections  (block tile 128x128)
  dim3 gD(Mtok / 128, Dd / 128);
  gemm_bf16_kernel<<<gD, tb, 0, stream>>>(zb, WqT, Dd, Dd, EP_QK, nullptr, nullptr,
                                          nullptr, Qbuf);
  gemm_bf16_kernel<<<gD, tb, 0, stream>>>(zb, WkT, Dd, Dd, EP_QK, nullptr, nullptr,
                                          nullptr, Kbuf);
  gemm_bf16_kernel<<<gD, tb, 0, stream>>>(zb, WvT, Dd, Dd, EP_VT, nullptr, nullptr,
                                          nullptr, Vtb);

  // causal flash attention
  attn_kernel<<<dim3(Ss / 16 / 8, Hh, Bb), tb, 0, stream>>>(Qbuf, Kbuf, Vtb, attO);

  // z_add = x + attO @ Wo + bo   (fp32)
  gemm_bf16_kernel<<<gD, tb, 0, stream>>>(attO, WoT, Dd, Dd, EP_ADDRES, bo, x,
                                          zadd, nullptr);

  // zn = LN2(z_add) (bf16)
  ln_kernel<<<Mtok, tb, 0, stream>>>(zadd, g2, be2, znb);

  // h = gelu(zn @ W1 + b1) (bf16)
  dim3 gF(Mtok / 128, Ff / 128);
  gemm_bf16_kernel<<<gF, tb, 0, stream>>>(znb, W1T, Ff, Dd, EP_GELU, b1, nullptr,
                                          nullptr, hbuf);

  // out = z_add + h @ W2 + b2 (fp32)
  gemm_bf16_kernel<<<gD, tb, 0, stream>>>(hbuf, W2T, Dd, Ff, EP_ADDRES, b2, zadd,
                                          (float*)d_out, nullptr);
}
